// VanillaMultiHeadAttention_25701084299730
// MI455X (gfx1250) — matrix operand order repaired and checked by reading, not compiled, not run on hardware
//
#include <hip/hip_runtime.h>

#define L_SEQ   4096
#define DMODEL  1024
#define N_HEADS 16
#define D_HEAD  64
#define SM_SCALE 0.125f  // 1/sqrt(64)

typedef __bf16 bf16_t;
typedef __attribute__((ext_vector_type(16))) __bf16 bf16x16;
typedef __attribute__((ext_vector_type(8)))  float  f32x8;
typedef __attribute__((ext_vector_type(4)))  int    int4v;

#if defined(__gfx1250__) && __has_builtin(__builtin_amdgcn_global_load_async_to_lds_b128) && __has_builtin(__builtin_amdgcn_s_wait_asynccnt)
#define HAVE_ASYNC_LDS 1
#else
#define HAVE_ASYNC_LDS 0
#endif

union Frag {
  unsigned int u[8];
  bf16x16 v;
};

static __device__ __forceinline__ f32x8 wmma_bf16(const Frag& a, const Frag& b, f32x8 c) {
  // D = A(16x32 bf16) * B(32x16 bf16) + C(16x16 f32)
  return __builtin_amdgcn_wmma_f32_16x16x32_bf16(false, a.v, false, b.v,
                                                 (short)0, c, false, false);
}

// 16-byte global -> LDS copy, async (ASYNCcnt) when available.
static __device__ __forceinline__ void copy16_g2l(const bf16_t* __restrict__ gsrc,
                                                  bf16_t* lds_dst) {
#if HAVE_ASYNC_LDS
  __builtin_amdgcn_global_load_async_to_lds_b128(
      (__attribute__((address_space(1))) int4v*)gsrc,
      (__attribute__((address_space(3))) int4v*)lds_dst, 0, 0);
#else
  *(uint4*)lds_dst = *(const uint4*)gsrc;
#endif
}

static __device__ __forceinline__ void wait_g2l() {
#if HAVE_ASYNC_LDS
  __builtin_amdgcn_s_wait_asynccnt(0);
#endif
}

// ---------------------------------------------------------------------------
// fp32 -> bf16 conversion (grid-stride)
// ---------------------------------------------------------------------------
__global__ void cvt_f32_bf16(const float* __restrict__ src,
                             bf16_t* __restrict__ dst, int n) {
  for (int i = blockIdx.x * blockDim.x + threadIdx.x; i < n;
       i += gridDim.x * blockDim.x) {
    dst[i] = (bf16_t)src[i];
  }
}

// ---------------------------------------------------------------------------
// WMMA GEMM:  C[m][n] = sum_k A[m][k] * Bt[n][k]
//   A : bf16 [4096][1024] row-major, Bt : bf16 [1024][1024] row-major (== W)
//   mode 0: bf16 -> out[(n/64)][m][n%64]     (Q, K head-major)
//   mode 2: bf16 -> out[(n/64)][n%64][m]     (V transposed)
//   mode 3: f32  -> outF[m][n]               (final projection)
// Block: 256 threads = 8 waves. Tile M=256 (32/wave), N=64, K-step 32.
// B tile double-buffered in LDS via async-to-LDS; A direct from global.
// One barrier per K-step: the top barrier (after s_wait_asynccnt) already
// orders "all waves done reading buf" before "any wave re-stages buf".
// ---------------------------------------------------------------------------
#define LDSB_STRIDE 72   // bf16 elems per row: 144 B, 16B-aligned, conflict-free

__global__ __launch_bounds__(256)
void gemm_bf16(const bf16_t* __restrict__ A, const bf16_t* __restrict__ Bt,
               bf16_t* __restrict__ outB, float* __restrict__ outF, int mode) {
  __shared__ bf16_t ldsB[2][64 * LDSB_STRIDE];   // 2 x 9 KB

  const int tid   = threadIdx.x;
  const int lane  = tid & 31;
  const int wave  = tid >> 5;
  const int l16   = lane & 15;
  const int g     = lane >> 4;                   // lane group (0 or 1)
  const int mbase = blockIdx.x * 256 + wave * 32;
  const int nbase = blockIdx.y * 64;

  // Cooperative B-tile stage: 64 rows x 32 cols bf16 = 256 x 16B chunks.
  const int srow = tid >> 2;
  const int scol = (tid & 3) * 8;
  const bf16_t* sgsrc = Bt + (size_t)(nbase + srow) * DMODEL + scol;
  bf16_t* const sldst0 = &ldsB[0][srow * LDSB_STRIDE + scol];
  bf16_t* const sldst1 = &ldsB[1][srow * LDSB_STRIDE + scol];

  const f32x8 zero = {0.f, 0.f, 0.f, 0.f, 0.f, 0.f, 0.f, 0.f};
  f32x8 acc[2][4];
#pragma unroll
  for (int s2 = 0; s2 < 2; ++s2)
#pragma unroll
    for (int t = 0; t < 4; ++t) acc[s2][t] = zero;

  copy16_g2l(sgsrc, sldst0);                     // prologue stage, k0 = 0
  int buf = 0;

  for (int k0 = 0; k0 < DMODEL; k0 += 32) {
    wait_g2l();
    __syncthreads();                             // staged tile visible; prev reads done
    if (k0 + 32 < DMODEL)                        // stage next tile into other buffer
      copy16_g2l(sgsrc + (k0 + 32), buf ? sldst0 : sldst1);

    // A fragments: two 16-row strips, K layout 2*j + g*16 (+h)
    Frag af[2];
#pragma unroll
    for (int s2 = 0; s2 < 2; ++s2) {
      const bf16_t* arow =
          A + (size_t)(mbase + s2 * 16 + l16) * DMODEL + k0 + g * 16;
#pragma unroll
      for (int j = 0; j < 8; ++j)
        af[s2].u[j] = *(const unsigned int*)(arow + 2 * j);
    }

    // All B fragments first (in-order DS returns -> staggered dscnt waits)
    Frag bfr[4];
#pragma unroll
    for (int t = 0; t < 4; ++t) {
      const bf16_t* brow = &ldsB[buf][(t * 16 + l16) * LDSB_STRIDE + g * 16];
#pragma unroll
      for (int v = 0; v < 8; ++v)
        bfr[t].u[v] = *(const unsigned int*)(brow + 2 * v);
    }
#pragma unroll
    for (int t = 0; t < 4; ++t) {
      acc[0][t] = wmma_bf16(af[0], bfr[t], acc[0][t]);
      acc[1][t] = wmma_bf16(af[1], bfr[t], acc[1][t]);
    }

    buf ^= 1;
  }

  // Store: C/D layout -> M = r + 8*g, N = l16
#pragma unroll
  for (int s2 = 0; s2 < 2; ++s2) {
#pragma unroll
    for (int t = 0; t < 4; ++t) {
#pragma unroll
      for (int r = 0; r < 8; ++r) {
        const int m = mbase + s2 * 16 + r + 8 * g;
        const int n = nbase + t * 16 + l16;
        const float val = acc[s2][t][r];
        if (mode == 3) {
          outF[(size_t)m * DMODEL + n] = val;
        } else if (mode == 2) {
          const int h = n >> 6, d = n & 63;          // V^T : [H][64][L]
          outB[((size_t)h * D_HEAD + d) * L_SEQ + m] = (bf16_t)val;
        } else {
          const int h = n >> 6, d = n & 63;          // Q/K : [H][L][64]
          outB[((size_t)h * L_SEQ + m) * D_HEAD + d] = (bf16_t)val;
        }
      }
    }
  }
}

// ---------------------------------------------------------------------------
// Flash attention (online softmax), one head per blockIdx.y.
// Block: 128 threads = 4 waves; each wave owns 16 query rows; key tile = 64.
//   Q,K : bf16 [H][L][64]   V^T : bf16 [H][64][L]
//   Out : bf16 [L][1024]  at column h*64+d  (A-layout for final projection)
// K and V tiles (8 KB each) are shared by all 4 waves: staged cooperatively
// in double-buffered LDS via async-to-LDS, one barrier per key tile.
// P transpose uses a per-wave private LDS region (in-order DS, no barrier).
// ---------------------------------------------------------------------------
__global__ __launch_bounds__(128)
void flash_attn(const bf16_t* __restrict__ Q, const bf16_t* __restrict__ K,
                const bf16_t* __restrict__ Vt, bf16_t* __restrict__ Oout) {
  __shared__ bf16_t ldsK[2][64 * LDSB_STRIDE];   // key tile   [key][d]
  __shared__ bf16_t ldsV[2][64 * LDSB_STRIDE];   // value tile [d][key]
  __shared__ bf16_t ldsP[4][16][64];             // per-wave P staging

  const int tid  = threadIdx.x;
  const int lane = tid & 31;
  const int wave = tid >> 5;
  const int l16  = lane & 15;
  const int g    = lane >> 4;
  const int h    = blockIdx.y;
  const int q0   = blockIdx.x * 64 + wave * 16;

  const bf16_t* Qh = Q  + (size_t)h * L_SEQ * D_HEAD;
  const bf16_t* Kh = K  + (size_t)h * L_SEQ * D_HEAD;
  const bf16_t* Vh = Vt + (size_t)h * D_HEAD * L_SEQ;

  // Q A-fragments for the wave's 16 rows: two frags cover d = 0..31, 32..63
  Frag qf[2];
  {
    const bf16_t* qrow = Qh + (size_t)(q0 + l16) * D_HEAD + g * 16;
#pragma unroll
    for (int p = 0; p < 2; ++p)
#pragma unroll
      for (int j = 0; j < 8; ++j) {
        const int k = p * 32 + 2 * j;
        qf[p].u[j] = *(const unsigned int*)(qrow + k);
      }
  }

  const f32x8 zero = {0.f, 0.f, 0.f, 0.f, 0.f, 0.f, 0.f, 0.f};
  f32x8 oacc[4];
#pragma unroll
  for (int t = 0; t < 4; ++t) oacc[t] = zero;
  float m_i[8], l_i[8];
#pragma unroll
  for (int r = 0; r < 8; ++r) { m_i[r] = -1e30f; l_i[r] = 0.f; }

  // Cooperative K/V tile staging: 64 rows x 64 cols bf16 = 512 chunks each;
  // 128 threads -> 4 chunks per tile per thread.
#define STAGE_KV(ktile, b)                                                    \
  {                                                                           \
    _Pragma("unroll")                                                         \
    for (int i = 0; i < 4; ++i) {                                             \
      const int c   = tid * 4 + i;                                            \
      const int row = c >> 3;                                                 \
      const int col = (c & 7) * 8;                                            \
      copy16_g2l(Kh + (size_t)((ktile) + row) * D_HEAD + col,                 \
                 &ldsK[b][row * LDSB_STRIDE + col]);                          \
      copy16_g2l(Vh + (size_t)row * L_SEQ + (ktile) + col,                    \
                 &ldsV[b][row * LDSB_STRIDE + col]);                          \
    }                                                                         \
  }

  STAGE_KV(0, 0);
  int buf = 0;

  for (int kt = 0; kt < L_SEQ; kt += 64) {
    wait_g2l();
    __syncthreads();                             // tiles visible; prev reads done
    if (kt + 64 < L_SEQ) STAGE_KV(kt + 64, buf ^ 1);

    // ---- scores: S(16x64) = Q(16x64) * K^T, 4 tiles of 16x16 ----
    f32x8 s[4];
#pragma unroll
    for (int t = 0; t < 4; ++t) {
      f32x8 c = zero;
#pragma unroll
      for (int p = 0; p < 2; ++p) {
        Frag bfr;  // B^T[n=key][k=d] == ldsK row-major
        const bf16_t* krow =
            &ldsK[buf][(t * 16 + l16) * LDSB_STRIDE + p * 32 + g * 16];
#pragma unroll
        for (int v = 0; v < 8; ++v)
          bfr.u[v] = *(const unsigned int*)(krow + 2 * v);
        c = wmma_bf16(qf[p], bfr, c);
      }
      s[t] = c;
    }

    // ---- online softmax across the 64 keys of this tile ----
#pragma unroll
    for (int r = 0; r < 8; ++r) {
      float mx = -1e30f;
#pragma unroll
      for (int t = 0; t < 4; ++t) mx = fmaxf(mx, s[t][r] * SM_SCALE);
#pragma unroll
      for (int msk = 1; msk < 16; msk <<= 1)
        mx = fmaxf(mx, __shfl_xor(mx, msk, 32));
      const float mnew  = fmaxf(m_i[r], mx);
      const float alpha = __expf(m_i[r] - mnew);
      float rs = 0.f;
#pragma unroll
      for (int t = 0; t < 4; ++t) {
        const float p = __expf(s[t][r] * SM_SCALE - mnew);
        s[t][r] = p;
        rs += p;
      }
#pragma unroll
      for (int msk = 1; msk < 16; msk <<= 1)
        rs += __shfl_xor(rs, msk, 32);
      l_i[r] = l_i[r] * alpha + rs;
      m_i[r] = mnew;
#pragma unroll
      for (int t = 0; t < 4; ++t) oacc[t][r] *= alpha;
    }

    // ---- transpose P through this wave's private LDS region ----
#pragma unroll
    for (int t = 0; t < 4; ++t)
#pragma unroll
      for (int r = 0; r < 8; ++r)
        ldsP[wave][r + 8 * g][t * 16 + l16] = (bf16_t)s[t][r];

    Frag pf[2];
    {
      const bf16_t* prow = &ldsP[wave][l16][0];
#pragma unroll
      for (int p = 0; p < 2; ++p)
#pragma unroll
        for (int j = 0; j < 8; ++j) {
          const int k = p * 32 + 2 * j + g * 16;
          pf[p].u[j] = *(const unsigned int*)(prow + k);
        }
    }

    // ---- O(16x64) += P(16x64) * V(64x64-tile) ----
#pragma unroll
    for (int t = 0; t < 4; ++t) {
#pragma unroll
      for (int p = 0; p < 2; ++p) {
        Frag bfr;  // B^T[n=d][k=key] == ldsV row-major
        const bf16_t* vrow =
            &ldsV[buf][(t * 16 + l16) * LDSB_STRIDE + p * 32 + g * 16];
#pragma unroll
        for (int v = 0; v < 8; ++v)
          bfr.u[v] = *(const unsigned int*)(vrow + 2 * v);
        oacc[t] = wmma_bf16(pf[p], bfr, oacc[t]);
      }
    }

    buf ^= 1;
  }
#undef STAGE_KV

  // ---- epilogue: normalize and store bf16 [L][1024] ----
#pragma unroll
  for (int r = 0; r < 8; ++r) {
    const float inv = 1.0f / l_i[r];
    const int m = q0 + r + 8 * g;
#pragma unroll
    for (int t = 0; t < 4; ++t) {
      const int col = h * D_HEAD + t * 16 + l16;
      Oout[(size_t)m * DMODEL + col] = (bf16_t)(oacc[t][r] * inv);
    }
  }
}

// ---------------------------------------------------------------------------
extern "C" void kernel_launch(void* const* d_in, const int* in_sizes, int n_in,
                              void* d_out, int out_size, void* d_ws, size_t ws_size,
                              hipStream_t stream) {
  const float* x  = (const float*)d_in[0];
  const float* Wq = (const float*)d_in[1];
  const float* Wk = (const float*)d_in[2];
  const float* Wv = (const float*)d_in[3];
  const float* Wo = (const float*)d_in[4];
  float* out = (float*)d_out;

  const size_t nX = (size_t)L_SEQ * DMODEL;   // 4M
  const size_t nW = (size_t)DMODEL * DMODEL;  // 1M

  bf16_t* xb  = (bf16_t*)d_ws;
  bf16_t* wqb = xb  + nX;
  bf16_t* wkb = wqb + nW;
  bf16_t* wvb = wkb + nW;
  bf16_t* wob = wvb + nW;
  bf16_t* Qh  = wob + nW;   // [16][4096][64]
  bf16_t* Kh  = Qh  + nX;
  bf16_t* Vt  = Kh  + nX;   // [16][64][4096]
  bf16_t* Ao  = Vt  + nX;   // [4096][1024]

  cvt_f32_bf16<<<1024, 256, 0, stream>>>(x,  xb,  (int)nX);
  cvt_f32_bf16<<<512,  256, 0, stream>>>(Wq, wqb, (int)nW);
  cvt_f32_bf16<<<512,  256, 0, stream>>>(Wk, wkb, (int)nW);
  cvt_f32_bf16<<<512,  256, 0, stream>>>(Wv, wvb, (int)nW);
  cvt_f32_bf16<<<512,  256, 0, stream>>>(Wo, wob, (int)nW);

  dim3 gg(L_SEQ / 256, DMODEL / 64);
  gemm_bf16<<<gg, 256, 0, stream>>>(xb, wqb, Qh, nullptr, 0);
  gemm_bf16<<<gg, 256, 0, stream>>>(xb, wkb, Kh, nullptr, 0);
  gemm_bf16<<<gg, 256, 0, stream>>>(xb, wvb, Vt, nullptr, 2);

  flash_attn<<<dim3(L_SEQ / 64, N_HEADS), 128, 0, stream>>>(Qh, Kh, Vt, Ao);

  gemm_bf16<<<gg, 256, 0, stream>>>(Ao, wob, nullptr, out, 3);
}
